// GNNLayer_40303973105841
// MI455X (gfx1250) — matrix operand order repaired and checked by reading, not compiled, not run on hardware
//
#include <hip/hip_runtime.h>

// ---------------------------------------------------------------------------
// GNN layer for MI455X (gfx1250, wave32, WMMA).
//
// msg  = relu(concat(node[src], edge) @ Wm.T + bm)        [E,128], K=192
// h    = segment_sum(msg, dst)                            [N,128]
// out  = relu(concat(node, h) @ Wa.T + ba)                [N,128], K=256
//
// f32 math emulated via split-bf16 (hi+lo) on v_wmma_f32_16x16x32_bf16:
// hi*hi + hi*lo + lo*hi accumulated in f32  (~f32 accuracy, 3x bf16 cost).
// node_features pre-split to bf16 hi/lo once (40K rows) so the 640K-edge
// gather loop does zero conversion VALU for node k-tiles.
// B fragments staged into LDS via global_load_async_to_lds_b128 (ASYNCcnt).
// Each wave computes TWO 16-row M-tiles so every B ds_load feeds 6 WMMAs
// (halves LDS read traffic vs one M-tile per wave).
// ---------------------------------------------------------------------------

#define N_NODES 40000
#define N_EDGES 640000
#define NODE_D  128
#define EDGE_D  64
#define OUT_D   128
#define K_MSG   (NODE_D + EDGE_D)   // 192 -> 6 k-tiles of 32
#define K_APP   (OUT_D + NODE_D)    // 256 -> 8 k-tiles of 32
#define NT      (OUT_D / 16)        // 8 n-tiles

typedef __attribute__((ext_vector_type(16))) __bf16 v16bf;
typedef __attribute__((ext_vector_type(8)))  float  v8f;

union Frag {
  unsigned short s[16];
  uint4          q[2];
  v16bf          v;
};

static __device__ __forceinline__ unsigned short bfbits(__bf16 b) {
  union { __bf16 b; unsigned short u; } x;
  x.b = b;
  return x.u;
}

static __device__ __forceinline__ v8f wmma_bf16(v16bf a, v16bf b, v8f c) {
  // (neg_a, A, neg_b, B, c_mod, C, reuse_a, reuse_b)
  return __builtin_amdgcn_wmma_f32_16x16x32_bf16(false, a, false, b,
                                                 (short)0, c, false, false);
}

// f32 run -> hi/lo bf16 halves of an A fragment (native v_cvt bf16 path).
static __device__ __forceinline__ void split8_f32(const float* __restrict__ p,
                                                  Frag& hi, Frag& lo, int base) {
  float4 a = reinterpret_cast<const float4*>(p)[0];
  float4 b = reinterpret_cast<const float4*>(p)[1];
  float f[8] = {a.x, a.y, a.z, a.w, b.x, b.y, b.z, b.w};
#pragma unroll
  for (int h = 0; h < 8; ++h) {
    __bf16 hb = (__bf16)f[h];
    hi.v[base + h] = hb;
    lo.v[base + h] = (__bf16)(f[h] - (float)hb);
  }
}

// Pre-split bf16 run -> A fragment halves: pure 16B loads, no VALU.
static __device__ __forceinline__ void load8_bf(const unsigned short* __restrict__ hiRow,
                                                const unsigned short* __restrict__ loRow,
                                                int c, Frag& hi, Frag& lo, int qi) {
  hi.q[qi] = *reinterpret_cast<const uint4*>(hiRow + c);
  lo.q[qi] = *reinterpret_cast<const uint4*>(loRow + c);
}

// Stage n_uint4*16 bytes of B fragments into LDS with async DMA (ASYNCcnt).
// LDS operand = low 32 bits of the flat shared address (aperture offset).
static __device__ __forceinline__ void stage_lds(uint4* ldsbuf,
                                                 const uint4* __restrict__ g,
                                                 int n_uint4, int tid) {
  for (int i = tid; i < n_uint4; i += 256) {
    unsigned loff = (unsigned)(size_t)(const void*)(ldsbuf + i);
    const uint4* ga = g + i;
    asm volatile("global_load_async_to_lds_b128 %0, %1, off"
                 :: "v"(loff), "v"(ga) : "memory");
  }
  asm volatile("s_wait_asynccnt 0" ::: "memory");
}

// ---------------------------------------------------------------------------
// Pack W [OUT_D x K] (row-major, out = A @ W.T) into per-lane WMMA B-matrix
// fragments, split hi/lo bf16.  B[k][n] = W[ntile*16+n][k].
// B layout (16-bit, 32x16): lanes 0-15 -> K={0..7,16..23}, lanes 16-31 ->
// K={8..15,24..31} (dword r = 2 consecutive K).  Linear: dword idx = frag*256 + lane*8 + r.
// ---------------------------------------------------------------------------
__global__ __launch_bounds__(256) void gnn_pack_b(const float* __restrict__ W, int K,
                                                  unsigned* __restrict__ hi,
                                                  unsigned* __restrict__ lo) {
  int idx    = blockIdx.x * 256 + threadIdx.x;   // grid sized exactly
  int f      = idx >> 8;
  int within = idx & 255;
  int lane   = within >> 3;
  int r      = within & 7;
  int kt = f >> 3, nt = f & 7;
  int n_global = nt * 16 + (lane & 15);
  int kbase    = kt * 32 + ((lane & 16) ? 8 : 0);
  int k0       = kbase + 2 * r + ((r >= 4) ? 8 : 0);
  float w0 = W[n_global * K + k0];
  float w1 = W[n_global * K + k0 + 1];
  __bf16 h0 = (__bf16)w0, h1 = (__bf16)w1;
  hi[idx] = (unsigned)bfbits(h0) | ((unsigned)bfbits(h1) << 16);
  __bf16 l0 = (__bf16)(w0 - (float)h0);
  __bf16 l1 = (__bf16)(w1 - (float)h1);
  lo[idx] = (unsigned)bfbits(l0) | ((unsigned)bfbits(l1) << 16);
}

// Split node_features into bf16 hi/lo arrays (same [row*128+col] indexing).
__global__ __launch_bounds__(256) void gnn_split_nodes(const float* __restrict__ nf,
                                                       unsigned short* __restrict__ hi,
                                                       unsigned short* __restrict__ lo) {
  int g = blockIdx.x * 256 + threadIdx.x;        // exact grid, 4 elems/thread
  float4 v = reinterpret_cast<const float4*>(nf)[g];
  float f[4] = {v.x, v.y, v.z, v.w};
  union { unsigned short s[4]; uint2 u; } H, L;
#pragma unroll
  for (int h = 0; h < 4; ++h) {
    __bf16 hb = (__bf16)f[h];
    H.s[h] = bfbits(hb);
    L.s[h] = bfbits((__bf16)(f[h] - (float)hb));
  }
  reinterpret_cast<uint2*>(hi)[g] = H.u;
  reinterpret_cast<uint2*>(lo)[g] = L.u;
}

__global__ __launch_bounds__(256) void gnn_zero(uint4* __restrict__ p) {
  p[blockIdx.x * 256 + threadIdx.x] = make_uint4(0u, 0u, 0u, 0u);  // exact grid
}

// ---------------------------------------------------------------------------
// Edge kernel: 8 waves/block, 32 edges per wave (two 16-row M-tiles),
// 256 edges/block (2500 blocks). B (hi+lo, 96 KB) async-staged in LDS once
// per block; each B fragment ds_load feeds 6 WMMAs (2 M-tiles x 3 products).
// k-tiles 0-3: node part (pre-split bf16 loads); k-tiles 4-5: edge part
// (f32 -> bf16 split on the fly, read-once data).
// relu+bias then f32 atomic scatter into h_neigh.
// ---------------------------------------------------------------------------
__global__ __launch_bounds__(256) void gnn_msg(const unsigned short* __restrict__ nodeHi,
                                               const unsigned short* __restrict__ nodeLo,
                                               const float* __restrict__ edgef,
                                               const int*   __restrict__ src,
                                               const int*   __restrict__ dst,
                                               const uint4* __restrict__ bpack,
                                               const float* __restrict__ bias,
                                               float* __restrict__ hneigh) {
  extern __shared__ uint4 lds[];
  const int tid = threadIdx.x;
  stage_lds(lds, bpack, 6144, tid);                 // 96 KB: hi|lo
  __syncthreads();
  const uint4* bhi = lds;
  const uint4* blo = lds + 3072;

  const int lane = tid & 31, wave = tid >> 5;
  const int e0 = blockIdx.x * 256 + wave * 32;      // 32 edges per wave
  const int mr = lane & 15;              // A-matrix row this lane feeds
  const int hiHalf = (lane >> 4) & 1;

  const int em0 = e0 + mr;
  const int em1 = e0 + 16 + mr;
  const size_t srow0 = (size_t)src[em0] * NODE_D;
  const size_t srow1 = (size_t)src[em1] * NODE_D;
  const unsigned short* hiRow[2] = {nodeHi + srow0, nodeHi + srow1};
  const unsigned short* loRow[2] = {nodeLo + srow0, nodeLo + srow1};
  const float* edgeRow[2] = {edgef + (size_t)em0 * EDGE_D,
                             edgef + (size_t)em1 * EDGE_D};

  v8f acc[2][NT];
#pragma unroll
  for (int m = 0; m < 2; ++m)
#pragma unroll
    for (int n = 0; n < NT; ++n)
#pragma unroll
      for (int r = 0; r < 8; ++r) acc[m][n][r] = 0.0f;

#pragma unroll
  for (int t = 0; t < 6; ++t) {
    Frag ah[2], al[2];
    // A layout 16x32 bf16: lanes0-15 K runs {0..7,16..23}, lanes16-31 {8..15,24..31}
    int c0 = t * 32 + hiHalf * 8;
#pragma unroll
    for (int m = 0; m < 2; ++m) {
      if (t < 4) {                        // node part: direct bf16 loads
        load8_bf(hiRow[m], loRow[m], c0,      ah[m], al[m], 0);
        load8_bf(hiRow[m], loRow[m], c0 + 16, ah[m], al[m], 1);
      } else {                            // edge part: convert on the fly
        split8_f32(edgeRow[m] + (c0 - NODE_D),      ah[m], al[m], 0);
        split8_f32(edgeRow[m] + (c0 + 16 - NODE_D), ah[m], al[m], 8);
      }
    }
#pragma unroll
    for (int n = 0; n < NT; ++n) {
      int fb = ((t * NT + n) * 32 + lane) * 2;
      Frag bh, bl;
      bh.q[0] = bhi[fb]; bh.q[1] = bhi[fb + 1];
      bl.q[0] = blo[fb]; bl.q[1] = blo[fb + 1];
#pragma unroll
      for (int m = 0; m < 2; ++m) {
        acc[m][n] = wmma_bf16(ah[m].v, bh.v, acc[m][n]);
        acc[m][n] = wmma_bf16(ah[m].v, bl.v, acc[m][n]);
        acc[m][n] = wmma_bf16(al[m].v, bh.v, acc[m][n]);
      }
    }
  }

  // C/D layout: element r -> row (r + 8*hiHalf), lane&15 -> column in tile.
#pragma unroll
  for (int m = 0; m < 2; ++m) {
    int dstIdx[8];
#pragma unroll
    for (int r = 0; r < 8; ++r) dstIdx[r] = dst[e0 + m * 16 + hiHalf * 8 + r];
#pragma unroll
    for (int n = 0; n < NT; ++n) {
      int col = n * 16 + mr;
      float bb = bias[col];
#pragma unroll
      for (int r = 0; r < 8; ++r) {
        float v = acc[m][n][r] + bb;
        v = v > 0.0f ? v : 0.0f;
        atomicAdd(hneigh + (size_t)dstIdx[r] * OUT_D + col, v);
      }
    }
  }
}

// ---------------------------------------------------------------------------
// Apply kernel: out = relu(concat(node, h_neigh) @ Wa.T + ba), K=256.
// 128 KB async-staged LDS for B hi/lo; 32 nodes per wave (two M-tiles).
// k-tiles 0-3 from pre-split nodes, k-tiles 4-7 from h_neigh (f32 split).
// ---------------------------------------------------------------------------
__global__ __launch_bounds__(256) void gnn_apply(const unsigned short* __restrict__ nodeHi,
                                                 const unsigned short* __restrict__ nodeLo,
                                                 const float* __restrict__ hneigh,
                                                 const uint4* __restrict__ bpack,
                                                 const float* __restrict__ bias,
                                                 float* __restrict__ out) {
  extern __shared__ uint4 lds[];
  const int tid = threadIdx.x;
  stage_lds(lds, bpack, 8192, tid);                 // 128 KB: hi|lo
  __syncthreads();
  const uint4* bhi = lds;
  const uint4* blo = lds + 4096;

  const int lane = tid & 31, wave = tid >> 5;
  const int node0 = blockIdx.x * 256 + wave * 32;   // 32 nodes per wave
  if (node0 >= N_NODES) return;  // 40000 % 32 == 0: whole wave in or out
  const int mr = lane & 15;
  const int hiHalf = (lane >> 4) & 1;

  const size_t nrow0 = (size_t)(node0 + mr) * NODE_D;
  const size_t nrow1 = (size_t)(node0 + 16 + mr) * NODE_D;
  const unsigned short* hiRow[2] = {nodeHi + nrow0, nodeHi + nrow1};
  const unsigned short* loRow[2] = {nodeLo + nrow0, nodeLo + nrow1};
  const float* hRow[2] = {hneigh + (size_t)(node0 + mr) * OUT_D,
                          hneigh + (size_t)(node0 + 16 + mr) * OUT_D};

  v8f acc[2][NT];
#pragma unroll
  for (int m = 0; m < 2; ++m)
#pragma unroll
    for (int n = 0; n < NT; ++n)
#pragma unroll
      for (int r = 0; r < 8; ++r) acc[m][n][r] = 0.0f;

#pragma unroll
  for (int t = 0; t < 8; ++t) {
    Frag ah[2], al[2];
    int c0 = t * 32 + hiHalf * 8;
#pragma unroll
    for (int m = 0; m < 2; ++m) {
      if (t < 4) {                        // node part: direct bf16 loads
        load8_bf(hiRow[m], loRow[m], c0,      ah[m], al[m], 0);
        load8_bf(hiRow[m], loRow[m], c0 + 16, ah[m], al[m], 1);
      } else {                            // aggregated part: convert on the fly
        split8_f32(hRow[m] + (c0 - NODE_D),      ah[m], al[m], 0);
        split8_f32(hRow[m] + (c0 + 16 - NODE_D), ah[m], al[m], 8);
      }
    }
#pragma unroll
    for (int n = 0; n < NT; ++n) {
      int fb = ((t * NT + n) * 32 + lane) * 2;
      Frag bh, bl;
      bh.q[0] = bhi[fb]; bh.q[1] = bhi[fb + 1];
      bl.q[0] = blo[fb]; bl.q[1] = blo[fb + 1];
#pragma unroll
      for (int m = 0; m < 2; ++m) {
        acc[m][n] = wmma_bf16(ah[m].v, bh.v, acc[m][n]);
        acc[m][n] = wmma_bf16(ah[m].v, bl.v, acc[m][n]);
        acc[m][n] = wmma_bf16(al[m].v, bh.v, acc[m][n]);
      }
    }
  }

#pragma unroll
  for (int m = 0; m < 2; ++m)
#pragma unroll
    for (int n = 0; n < NT; ++n) {
      int col = n * 16 + mr;
      float bb = bias[col];
#pragma unroll
      for (int r = 0; r < 8; ++r) {
        float v = acc[m][n][r] + bb;
        v = v > 0.0f ? v : 0.0f;
        out[(size_t)(node0 + m * 16 + hiHalf * 8 + r) * OUT_D + col] = v;
      }
    }
}

// ---------------------------------------------------------------------------
extern "C" void kernel_launch(void* const* d_in, const int* in_sizes, int n_in,
                              void* d_out, int out_size, void* d_ws, size_t ws_size,
                              hipStream_t stream) {
  const float* nodef = (const float*)d_in[0];
  const float* edgef = (const float*)d_in[1];
  const int*   src   = (const int*)d_in[2];
  const int*   dst   = (const int*)d_in[3];
  const float* Wm    = (const float*)d_in[4];
  const float* bm    = (const float*)d_in[5];
  const float* Wa    = (const float*)d_in[6];
  const float* ba    = (const float*)d_in[7];
  float* out = (float*)d_out;

  // Workspace layout (16B aligned):
  //  [0, 98304)            : packed W_msg  B-frags, hi(12288 dw) | lo(12288 dw)
  //  [98304, 229376)       : packed W_apply B-frags, hi(16384 dw) | lo(16384 dw)
  //  [229376, +20.48 MB)   : h_neigh accumulator [40000 x 128] f32
  //  [+20.48MB, +10.24 MB) : node_features hi bf16 [40000 x 128]
  //  [.., +10.24 MB)       : node_features lo bf16 [40000 x 128]
  char* ws = (char*)d_ws;
  unsigned* bmsg = (unsigned*)ws;
  unsigned* bapp = (unsigned*)(ws + 98304);
  float* hneigh  = (float*)(ws + 229376);
  unsigned short* nodeHi = (unsigned short*)(ws + 229376 + 20480000);
  unsigned short* nodeLo = (unsigned short*)(ws + 229376 + 20480000 + 10240000);

  gnn_pack_b<<<48, 256, 0, stream>>>(Wm, K_MSG, bmsg, bmsg + 12288);
  gnn_pack_b<<<64, 256, 0, stream>>>(Wa, K_APP, bapp, bapp + 16384);
  gnn_zero<<<5000, 256, 0, stream>>>((uint4*)hneigh);          // 5.12M floats
  gnn_split_nodes<<<5000, 256, 0, stream>>>(nodef, nodeHi, nodeLo);

  gnn_msg<<<N_EDGES / 256, 256, 96 * 1024, stream>>>(
      nodeHi, nodeLo, edgef, src, dst, (const uint4*)bmsg, bm, hneigh);

  gnn_apply<<<(N_NODES + 255) / 256, 256, 128 * 1024, stream>>>(
      nodeHi, nodeLo, hneigh, (const uint4*)bapp, ba, out);
}
